// WASMOptimizedLSTM_28630251995538
// MI455X (gfx1250) — compile-verified, matrix operand order checked by reading, not run on hardware
//
#include <hip/hip_runtime.h>
#include <hip/hip_bf16.h>

// ---------------------------------------------------------------------------
// LSTM stack (B=256, S=256, IN=10, H=256, L=4) + heads for MI455X (gfx1250).
// bf16 WMMA (v_wmma_f32_16x16x32_bf16) gate GEMMs with fused in-register LSTM
// cell update. Wavefront scheduling: one kernel launch per diagonal d = t + l
// (cells (t,l) on a diagonal are mutually independent), S+L-1 = 259 launches,
// up to 4 layer-cells (128 workgroups) in flight per launch.
// ---------------------------------------------------------------------------

#define Bdim   256
#define Sdim   256
#define INDIM  10
#define Hdim   256
#define H4     1024
#define Ldim   4
#define NCLS   5

typedef __attribute__((ext_vector_type(16))) __bf16 v16bf;
typedef __attribute__((ext_vector_type(8)))  float  v8f;

union FragU { uint4 u[2]; v16bf v; };

__device__ __forceinline__ unsigned short f2bf(float f) {
    unsigned int x = __float_as_uint(f);
    unsigned int r = x + 0x7fffu + ((x >> 16) & 1u);   // round-to-nearest-even
    return (unsigned short)(r >> 16);
}

// Load one 16x32 bf16 WMMA fragment from a row-major [rows][rowStride] bf16
// array. ISA layout (CDNA5 ISA): lane L holds row `row`, K pairs
// [hi*8, hi*8+8) in VGPR0..3 and [16+hi*8, ...) in VGPR4..7, hi = L/16.
// Both chunks are contiguous in row-major memory -> two b128 loads.
__device__ __forceinline__ v16bf load_frag(const unsigned short* base,
                                           int row, int rowStride, int kbase, int lane) {
    const int half = lane >> 4;
    const unsigned short* p = base + (size_t)row * rowStride + kbase + half * 8;
    FragU f;
    f.u[0] = *reinterpret_cast<const uint4*>(p);
    f.u[1] = *reinterpret_cast<const uint4*>(p + 16);
    return f.v;
}

__device__ __forceinline__ float fast_sigmoid(float x) {
    return 1.0f / (1.0f + __expf(-x));
}

// ---------------------------------------------------------------------------
// Wavefront kernel: processes all active LSTM cells (t, l) on diagonal d.
// blockIdx.z selects the layer (l = lmin + z, t = d - l).
// Grid: (B/16, H/128, nActiveLayers); block: 256 threads = 8 wave32s; each
// wave computes a 16x16 (batch x hidden) tile for all 4 gates, then applies
// the LSTM cell elementwise entirely in registers.
// ---------------------------------------------------------------------------
__global__ void __launch_bounds__(256)
lstm_wavefront_kernel(unsigned short* __restrict__ hbuf,   // [2][L][B][H] bf16 (parity double buffer)
                      const unsigned short* __restrict__ wb_hh, // [L][4H][H] bf16
                      const unsigned short* __restrict__ wb_ih, // [L-1][4H][H] bf16
                      const float* __restrict__ x,         // [B][S][IN]
                      const float* __restrict__ w_ih0,     // [4H][IN] f32
                      const float* __restrict__ b_ih_all,  // [L][4H]
                      const float* __restrict__ b_hh_all,  // [L][4H]
                      float* __restrict__ cbuf,            // [L][B][H] f32, in place
                      float* __restrict__ out_hidden,      // [B][S][H] f32
                      int d, int lmin)
{
    const int l  = lmin + blockIdx.z;
    const int t  = d - l;
    const int rp = t & 1;            // own-layer h_{t-1} parity
    const int wp = (t + 1) & 1;      // write parity (also layer-below h_t)

    const int BH = Bdim * Hdim;
    const unsigned short* h_prev = hbuf + ((size_t)rp * Ldim + l) * BH;
    unsigned short*       h_out  = hbuf + ((size_t)wp * Ldim + l) * BH;
    const unsigned short* h_in   = hbuf + ((size_t)wp * Ldim + (l - 1)) * BH; // valid iff l>0
    const unsigned short* w_hh   = wb_hh + (size_t)l * H4 * Hdim;
    const unsigned short* w_ih   = wb_ih + (size_t)(l - 1) * H4 * Hdim;       // valid iff l>0
    const float*          b_ih   = b_ih_all + (size_t)l * H4;
    const float*          b_hh   = b_hh_all + (size_t)l * H4;
    float*                c_st   = cbuf + (size_t)l * BH;
    const int is_l0 = (l == 0);
    const int top   = (l == Ldim - 1);

    const int tid  = threadIdx.x;
    const int lane = tid & 31;
    const int wave = tid >> 5;
    const int m0   = blockIdx.x * 16;
    const int n0   = blockIdx.y * 128 + wave * 16;
    const int col  = lane & 15;
    const int half = lane >> 4;
    const int n    = n0 + col;       // hidden-unit column owned by this lane
    const int mrow = m0 + col;       // A-fragment row for this lane

    __shared__ float xs[16 * INDIM];
    if (is_l0) {
        if (tid < 16 * INDIM) {
            int r = tid / INDIM, k = tid % INDIM;
            xs[tid] = x[(size_t)(m0 + r) * (Sdim * INDIM) + (size_t)t * INDIM + k];
        }
        __syncthreads();
    }

    // Accumulators for gates i, f, g, o — init with combined biases.
    v8f acc[4];
#pragma unroll
    for (int g = 0; g < 4; ++g) {
        float bv = b_ih[g * Hdim + n] + b_hh[g * Hdim + n];
#pragma unroll
        for (int r = 0; r < 8; ++r) acc[g][r] = bv;
    }

    // Recurrent term: h_{t-1} @ w_hh^T (K=256 -> 8 chunks). Issue all loads of
    // a chunk before its 4 WMMAs so the scheduler can batch them in clauses.
#pragma unroll
    for (int kt = 0; kt < Hdim; kt += 32) {
        v16bf a  = load_frag(h_prev, mrow, Hdim, kt, lane);
        v16bf b0 = load_frag(w_hh, 0 * Hdim + n, Hdim, kt, lane);
        v16bf b1 = load_frag(w_hh, 1 * Hdim + n, Hdim, kt, lane);
        v16bf b2 = load_frag(w_hh, 2 * Hdim + n, Hdim, kt, lane);
        v16bf b3 = load_frag(w_hh, 3 * Hdim + n, Hdim, kt, lane);
        acc[0] = __builtin_amdgcn_wmma_f32_16x16x32_bf16(false, a, false, b0, (short)0, acc[0], false, false);
        acc[1] = __builtin_amdgcn_wmma_f32_16x16x32_bf16(false, a, false, b1, (short)0, acc[1], false, false);
        acc[2] = __builtin_amdgcn_wmma_f32_16x16x32_bf16(false, a, false, b2, (short)0, acc[2], false, false);
        acc[3] = __builtin_amdgcn_wmma_f32_16x16x32_bf16(false, a, false, b3, (short)0, acc[3], false, false);
    }

    if (!is_l0) {
        // Inter-layer term: h_{l-1,t} @ w_ih^T (K=256)
#pragma unroll
        for (int kt = 0; kt < Hdim; kt += 32) {
            v16bf a  = load_frag(h_in, mrow, Hdim, kt, lane);
            v16bf b0 = load_frag(w_ih, 0 * Hdim + n, Hdim, kt, lane);
            v16bf b1 = load_frag(w_ih, 1 * Hdim + n, Hdim, kt, lane);
            v16bf b2 = load_frag(w_ih, 2 * Hdim + n, Hdim, kt, lane);
            v16bf b3 = load_frag(w_ih, 3 * Hdim + n, Hdim, kt, lane);
            acc[0] = __builtin_amdgcn_wmma_f32_16x16x32_bf16(false, a, false, b0, (short)0, acc[0], false, false);
            acc[1] = __builtin_amdgcn_wmma_f32_16x16x32_bf16(false, a, false, b1, (short)0, acc[1], false, false);
            acc[2] = __builtin_amdgcn_wmma_f32_16x16x32_bf16(false, a, false, b2, (short)0, acc[2], false, false);
            acc[3] = __builtin_amdgcn_wmma_f32_16x16x32_bf16(false, a, false, b3, (short)0, acc[3], false, false);
        }
    } else {
        // Layer-0 input term: x_t @ w_ih0^T, K=10 -> plain FMAs (f32).
        float wreg[4][INDIM];
#pragma unroll
        for (int g = 0; g < 4; ++g)
#pragma unroll
            for (int k = 0; k < INDIM; ++k)
                wreg[g][k] = w_ih0[(size_t)(g * Hdim + n) * INDIM + k];
#pragma unroll
        for (int r = 0; r < 8; ++r) {
            int mloc = r + half * 8;
#pragma unroll
            for (int k = 0; k < INDIM; ++k) {
                float xv = xs[mloc * INDIM + k];
#pragma unroll
                for (int g = 0; g < 4; ++g)
                    acc[g][r] = __builtin_fmaf(xv, wreg[g][k], acc[g][r]);
            }
        }
    }

    // In-register LSTM cell update. D layout: lane L, vgpr r -> (m = r+8*half,
    // n = L%16), so i/f/g/o for one (batch, hidden) element live in one lane.
#pragma unroll
    for (int r = 0; r < 8; ++r) {
        int m = m0 + r + half * 8;
        size_t idx = (size_t)m * Hdim + n;
        float si = fast_sigmoid(acc[0][r]);
        float sf = fast_sigmoid(acc[1][r]);
        float tg = tanhf(acc[2][r]);
        float so = fast_sigmoid(acc[3][r]);
        float c  = sf * c_st[idx] + si * tg;
        float h  = so * tanhf(c);
        c_st[idx] = c;
        h_out[idx] = f2bf(h);
        if (top) out_hidden[(size_t)m * (Sdim * Hdim) + (size_t)t * Hdim + n] = h;
    }
}

// ---------------------------------------------------------------------------
// Helpers: weight pack (f32 -> bf16), state zeroing, classifier heads.
// ---------------------------------------------------------------------------
__global__ void f32_to_bf16_kernel(const float* __restrict__ in,
                                   unsigned short* __restrict__ out, int n) {
    int i = blockIdx.x * 256 + threadIdx.x;
    if (i < n) out[i] = f2bf(in[i]);
}

__global__ void zero_state_kernel(unsigned short* __restrict__ h0,
                                  float* __restrict__ c, int n) {
    int i = blockIdx.x * 256 + threadIdx.x;
    if (i < n) { h0[i] = 0; c[i] = 0.0f; }
}

__global__ void heads_kernel(const float* __restrict__ hidden,  // [B][S][H]
                             const float* __restrict__ w_cls,   // [NCLS][H]
                             const float* __restrict__ b_cls,   // [NCLS]
                             const float* __restrict__ w_conf,  // [1][H]
                             const float* __restrict__ b_conf,  // [1]
                             float* __restrict__ patterns,      // [B][NCLS]
                             float* __restrict__ conf,          // [B]
                             float* __restrict__ final_hidden)  // [B][H]
{
    int b = blockIdx.x * blockDim.x + threadIdx.x;
    if (b >= Bdim) return;
    const float* last = hidden + ((size_t)b * Sdim + (Sdim - 1)) * Hdim;

    float logits[NCLS];
    for (int c5 = 0; c5 < NCLS; ++c5) {
        float s = b_cls[c5];
        for (int k = 0; k < Hdim; ++k) s += last[k] * w_cls[c5 * Hdim + k];
        logits[c5] = s;
    }
    float mx = logits[0];
    for (int c5 = 1; c5 < NCLS; ++c5) mx = fmaxf(mx, logits[c5]);
    float e[NCLS], den = 0.0f;
    for (int c5 = 0; c5 < NCLS; ++c5) { e[c5] = __expf(logits[c5] - mx); den += e[c5]; }
    for (int c5 = 0; c5 < NCLS; ++c5) patterns[b * NCLS + c5] = e[c5] / den;

    float s = b_conf[0];
    for (int k = 0; k < Hdim; ++k) s += last[k] * w_conf[k];
    conf[b] = fast_sigmoid(s);

    for (int k = 0; k < Hdim; ++k) final_hidden[(size_t)b * Hdim + k] = last[k];
}

// ---------------------------------------------------------------------------
// Host launcher
// ---------------------------------------------------------------------------
extern "C" void kernel_launch(void* const* d_in, const int* in_sizes, int n_in,
                              void* d_out, int out_size, void* d_ws, size_t ws_size,
                              hipStream_t stream) {
    (void)in_sizes; (void)n_in; (void)out_size; (void)ws_size;

    const float* x         = (const float*)d_in[0];   // [B][S][IN]
    const float* w_ih0     = (const float*)d_in[1];   // [4H][IN]
    const float* w_ih_rest = (const float*)d_in[2];   // [L-1][4H][H]
    const float* w_hh      = (const float*)d_in[3];   // [L][4H][H]
    const float* b_ih      = (const float*)d_in[4];   // [L][4H]
    const float* b_hh      = (const float*)d_in[5];   // [L][4H]
    const float* w_cls     = (const float*)d_in[6];   // [NCLS][H]
    const float* b_cls     = (const float*)d_in[7];   // [NCLS]
    const float* w_conf    = (const float*)d_in[8];   // [1][H]
    const float* b_conf    = (const float*)d_in[9];   // [1]

    float* out = (float*)d_out;
    float* out_patterns = out;                                     // B*NCLS
    float* out_conf     = out + Bdim * NCLS;                       // B
    float* out_hidden   = out_conf + Bdim;                         // B*S*H
    float* out_final    = out_hidden + (size_t)Bdim * Sdim * Hdim; // B*H

    // Workspace layout (all chunks multiple of 256 B, base is hipMalloc-aligned)
    char* ws = (char*)d_ws;
    size_t off = 0;
    unsigned short* wb_hh = (unsigned short*)(ws + off); off += (size_t)Ldim * H4 * Hdim * 2;       // 2 MB
    unsigned short* wb_ih = (unsigned short*)(ws + off); off += (size_t)(Ldim - 1) * H4 * Hdim * 2; // 1.5 MB
    unsigned short* hbuf  = (unsigned short*)(ws + off); off += (size_t)2 * Ldim * Bdim * Hdim * 2; // 1 MB
    float*          cbuf  = (float*)(ws + off);          off += (size_t)Ldim * Bdim * Hdim * 4;     // 1 MB

    // Pack weights to bf16 (row-major, layout unchanged).
    {
        int n1 = Ldim * H4 * Hdim;
        f32_to_bf16_kernel<<<(n1 + 255) / 256, 256, 0, stream>>>(w_hh, wb_hh, n1);
        int n2 = (Ldim - 1) * H4 * Hdim;
        f32_to_bf16_kernel<<<(n2 + 255) / 256, 256, 0, stream>>>(w_ih_rest, wb_ih, n2);
    }
    // Zero h (parity-0 buffer) and c state.
    {
        int n = Ldim * Bdim * Hdim;
        zero_state_kernel<<<(n + 255) / 256, 256, 0, stream>>>(hbuf, cbuf, n);
    }

    // Wavefront over diagonals d = t + l. All cells on a diagonal depend only
    // on the previous diagonal -> safe with stream-ordered launches.
    for (int d = 0; d < Sdim + Ldim - 1; ++d) {
        int lmin = d - (Sdim - 1); if (lmin < 0) lmin = 0;
        int lmax = (d < Ldim - 1) ? d : (Ldim - 1);
        dim3 grid(Bdim / 16, Hdim / 128, lmax - lmin + 1);
        lstm_wavefront_kernel<<<grid, 256, 0, stream>>>(
            hbuf, wb_hh, wb_ih, x, w_ih0, b_ih, b_hh,
            cbuf, out_hidden, d, lmin);
    }

    heads_kernel<<<(Bdim + 255) / 256, 256, 0, stream>>>(
        out_hidden, w_cls, b_cls, w_conf, b_conf,
        out_patterns, out_conf, out_final);
}
